// MambaBlock_26061861552218
// MI455X (gfx1250) — matrix operand order repaired and checked by reading, not compiled, not run on hardware
//
#include <hip/hip_runtime.h>
#include <hip/hip_bf16.h>
#include <math.h>

// ---------------- constants ----------------
#define D_MODEL   1024
#define D_STATE   16
#define D_CONV    4
#define D_INNER   2048
#define DT_RANK   64
#define BATCH     2
#define SEQ       1024
#define NTOK      (BATCH * SEQ)           // 2048 token rows
#define DBC_N     (DT_RANK + 2 * D_STATE) // 96

typedef _Float16 f16x8 __attribute__((ext_vector_type(8)));
typedef _Float16 v16h  __attribute__((ext_vector_type(16)));
typedef float    v8f   __attribute__((ext_vector_type(8)));

// ---------------- f32 -> f16 convert ----------------
__global__ __launch_bounds__(256) void cvt_f32_f16(const float* __restrict__ in,
                                                   _Float16* __restrict__ out, int n) {
    int i = blockIdx.x * 256 + threadIdx.x;
    if (i < n) out[i] = (_Float16)in[i];
}

// ---------------- LayerNorm (f32 in, f16 out) ----------------
__global__ __launch_bounds__(256) void layernorm_f16(const float* __restrict__ x,
                                                     const float* __restrict__ w,
                                                     const float* __restrict__ b,
                                                     _Float16* __restrict__ out, int cols) {
    __shared__ float s1[256], s2[256];
    const int row = blockIdx.x;
    const float* xr = x + (size_t)row * cols;
    float sum = 0.f, sq = 0.f;
    for (int c = threadIdx.x; c < cols; c += 256) { float v = xr[c]; sum += v; sq += v * v; }
    s1[threadIdx.x] = sum; s2[threadIdx.x] = sq;
    __syncthreads();
    for (int st = 128; st > 0; st >>= 1) {
        if (threadIdx.x < st) { s1[threadIdx.x] += s1[threadIdx.x + st]; s2[threadIdx.x] += s2[threadIdx.x + st]; }
        __syncthreads();
    }
    const float mu   = s1[0] / (float)cols;
    const float var  = s2[0] / (float)cols - mu * mu;
    const float rstd = rsqrtf(var + 1e-5f);
    _Float16* orow = out + (size_t)row * cols;
    for (int c = threadIdx.x; c < cols; c += 256)
        orow[c] = (_Float16)((xr[c] - mu) * rstd * w[c] + b[c]);
}

// ---------------- WMMA GEMM: C(MxN,f32) = A(MxK,f16) * W(NxK,f16)^T ----------------
// One wave per block; wave computes a 16 x (16*JN) output strip.
// Grid must exactly tile the output: grid.x = N/(16*JN), grid.y = M/16.
// K must be a multiple of 64. No branches in the k-loop; explicit 2-deep
// software pipeline (set0/set1) so loads overlap the WMMA pipe.
template <int JN>
__global__ __launch_bounds__(32) void gemm_wmma_f16(const _Float16* __restrict__ A, int lda,
                                                    const _Float16* __restrict__ W, int ldb,
                                                    float* __restrict__ C, int ldc, int K) {
    const int lane = threadIdx.x & 31;
    const int half = lane >> 4;   // 0: lanes 0-15, 1: lanes 16-31
    const int l16  = lane & 15;
    const int m0   = blockIdx.y * 16;
    const int n0   = blockIdx.x * (16 * JN);

    // A-fragment (16-bit A 16x32 layout, ISA 7.12.2): row = l16,
    // K chunks [kb + half*8 .. +7] and [kb + half*8 + 16 .. +23]
    const _Float16* Ap = A + (size_t)(m0 + l16) * lda + half * 8;
    // B-fragment: lane holds N = l16, K chunks [kb + half*8 .. +7] and [kb + half*8 + 16 .. +23]
    const _Float16* Wp[JN];
#pragma unroll
    for (int j = 0; j < JN; ++j)
        Wp[j] = W + (size_t)(n0 + j * 16 + l16) * ldb + half * 8;

    v8f acc[JN];
#pragma unroll
    for (int j = 0; j < JN; ++j) acc[j] = (v8f){};

    union F { v16h v; f16x8 h[2]; };
    F a0, a1, b0[JN], b1[JN];

    // preload set0 @ k=0
    a0.h[0] = *(const f16x8*)(Ap);
    a0.h[1] = *(const f16x8*)(Ap + 16);
#pragma unroll
    for (int j = 0; j < JN; ++j) {
        b0[j].h[0] = *(const f16x8*)(Wp[j]);
        b0[j].h[1] = *(const f16x8*)(Wp[j] + 16);
    }

    int kb = 0;
    for (; kb < K - 64; kb += 64) {
        // load set1 @ kb+32
        a1.h[0] = *(const f16x8*)(Ap + kb + 32);
        a1.h[1] = *(const f16x8*)(Ap + kb + 48);
#pragma unroll
        for (int j = 0; j < JN; ++j) {
            b1[j].h[0] = *(const f16x8*)(Wp[j] + kb + 32);
            b1[j].h[1] = *(const f16x8*)(Wp[j] + kb + 48);
        }
        // math on set0
#pragma unroll
        for (int j = 0; j < JN; ++j)
            acc[j] = __builtin_amdgcn_wmma_f32_16x16x32_f16(
                false, a0.v, false, b0[j].v, (short)0, acc[j], false, false);
        // load set0 @ kb+64
        a0.h[0] = *(const f16x8*)(Ap + kb + 64);
        a0.h[1] = *(const f16x8*)(Ap + kb + 80);
#pragma unroll
        for (int j = 0; j < JN; ++j) {
            b0[j].h[0] = *(const f16x8*)(Wp[j] + kb + 64);
            b0[j].h[1] = *(const f16x8*)(Wp[j] + kb + 80);
        }
        // math on set1
#pragma unroll
        for (int j = 0; j < JN; ++j)
            acc[j] = __builtin_amdgcn_wmma_f32_16x16x32_f16(
                false, a1.v, false, b1[j].v, (short)0, acc[j], false, false);
    }
    // tail: last 64 of K (kb == K-64 here)
    a1.h[0] = *(const f16x8*)(Ap + kb + 32);
    a1.h[1] = *(const f16x8*)(Ap + kb + 48);
#pragma unroll
    for (int j = 0; j < JN; ++j) {
        b1[j].h[0] = *(const f16x8*)(Wp[j] + kb + 32);
        b1[j].h[1] = *(const f16x8*)(Wp[j] + kb + 48);
    }
#pragma unroll
    for (int j = 0; j < JN; ++j)
        acc[j] = __builtin_amdgcn_wmma_f32_16x16x32_f16(
            false, a0.v, false, b0[j].v, (short)0, acc[j], false, false);
#pragma unroll
    for (int j = 0; j < JN; ++j)
        acc[j] = __builtin_amdgcn_wmma_f32_16x16x32_f16(
            false, a1.v, false, b1[j].v, (short)0, acc[j], false, false);

    // C/D layout: reg r -> row m0 + r + half*8, col n0 + j*16 + l16
    const int mrow = m0 + half * 8;
#pragma unroll
    for (int j = 0; j < JN; ++j) {
        float* Cp = C + (size_t)mrow * ldc + n0 + j * 16 + l16;
#pragma unroll
        for (int r = 0; r < 8; ++r) Cp[(size_t)r * ldc] = acc[j][r];
    }
}

// ---------------- depthwise causal conv (k=4) + SiLU ----------------
__global__ __launch_bounds__(256) void conv_silu(const float* __restrict__ xz,
                                                 const float* __restrict__ cw,
                                                 const float* __restrict__ cb,
                                                 float* __restrict__ u32,
                                                 _Float16* __restrict__ u16, int n) {
    int i = blockIdx.x * 256 + threadIdx.x;
    if (i >= n) return;
    const int d = i & (D_INNER - 1);
    const int t = (i >> 11) & (SEQ - 1);
    const int b = i >> 21;
    float acc = cb[d];
#pragma unroll
    for (int j = 0; j < D_CONV; ++j) {
        const int tt = t - (D_CONV - 1) + j;
        if (tt >= 0)
            acc += cw[d * D_CONV + j] * xz[((size_t)(b * SEQ + tt)) * (2 * D_INNER) + d];
    }
    const float s = acc / (1.f + expf(-acc));
    u32[i] = s;
    u16[i] = (_Float16)s;
}

// ---------------- softplus(delta_raw + bias), in place ----------------
__global__ __launch_bounds__(256) void softplus_bias(float* __restrict__ d,
                                                     const float* __restrict__ b, int n) {
    int i = blockIdx.x * 256 + threadIdx.x;
    if (i < n) {
        const float v = d[i] + b[i & (D_INNER - 1)];
        d[i] = (v > 20.f) ? v : log1pf(expf(v));
    }
}

// ---------------- selective scan ----------------
__global__ __launch_bounds__(256) void scan_kernel(const float* __restrict__ delta,
                                                   const float* __restrict__ u,
                                                   const float* __restrict__ dbc,
                                                   const float* __restrict__ A_log,
                                                   const float* __restrict__ Dp,
                                                   float* __restrict__ y) {
    const int g    = blockIdx.x * 256 + threadIdx.x;
    const int lane = g & 31;
    const int wave = g >> 5;              // 0..2047
    const int half = lane >> 4;
    const int n    = lane & 15;           // state index
    const int cg   = wave * 2 + half;     // 0..4095 global channel
    const int b    = cg >> 11;
    const int d    = cg & (D_INNER - 1);

    const float a  = -expf(A_log[d * D_STATE + n]);
    const float Dc = Dp[d];
    float h = 0.f;

    const float* drow = delta + (size_t)b * SEQ * D_INNER + d;
    const float* urow = u     + (size_t)b * SEQ * D_INNER + d;
    const float* brow = dbc   + (size_t)b * SEQ * DBC_N + DT_RANK + n;
    const float* crow = dbc   + (size_t)b * SEQ * DBC_N + DT_RANK + D_STATE + n;
    float*       yrow = y     + (size_t)b * SEQ * D_INNER + d;

    for (int t = 0; t < SEQ; ++t) {
        const float dt = drow[(size_t)t * D_INNER];
        const float ut = urow[(size_t)t * D_INNER];
        const float Bn = brow[(size_t)t * DBC_N];
        const float Cn = crow[(size_t)t * DBC_N];
        h = expf(dt * a) * h + dt * Bn * ut;
        float p = h * Cn;
        p += __shfl_xor(p, 1, 16);
        p += __shfl_xor(p, 2, 16);
        p += __shfl_xor(p, 4, 16);
        p += __shfl_xor(p, 8, 16);
        if (n == 0) yrow[(size_t)t * D_INNER] = p + Dc * ut;
    }
}

// ---------------- y * silu(z) -> f16 ----------------
__global__ __launch_bounds__(256) void gate_silu(const float* __restrict__ yscan,
                                                 const float* __restrict__ xz,
                                                 _Float16* __restrict__ y16, int n) {
    int i = blockIdx.x * 256 + threadIdx.x;
    if (i >= n) return;
    const int row = i >> 11;
    const int d   = i & (D_INNER - 1);
    const float z = xz[(size_t)row * (2 * D_INNER) + D_INNER + d];
    const float s = z / (1.f + expf(-z));
    y16[i] = (_Float16)(yscan[i] * s);
}

// ---------------- elementwise add (residual) ----------------
__global__ __launch_bounds__(256) void add_kernel(const float* __restrict__ a,
                                                  const float* __restrict__ b,
                                                  float* __restrict__ o, int n) {
    int i = blockIdx.x * 256 + threadIdx.x;
    if (i < n) o[i] = a[i] + b[i];
}

// ---------------- bias + exact GELU -> f16 ----------------
__global__ __launch_bounds__(256) void bias_gelu(const float* __restrict__ mid,
                                                 const float* __restrict__ b1,
                                                 _Float16* __restrict__ out, int n) {
    int i = blockIdx.x * 256 + threadIdx.x;
    if (i < n) {
        const float v = mid[i] + b1[i & (D_INNER - 1)];
        out[i] = (_Float16)(0.5f * v * (1.f + erff(v * 0.70710678118654752f)));
    }
}

// ---------------- final: out = h + y2 + b2 ----------------
__global__ __launch_bounds__(256) void final_add(const float* __restrict__ h,
                                                 const float* __restrict__ y2,
                                                 const float* __restrict__ b2,
                                                 float* __restrict__ out, int n) {
    int i = blockIdx.x * 256 + threadIdx.x;
    if (i < n) out[i] = h[i] + y2[i] + b2[i & (D_MODEL - 1)];
}

// =======================================================================
extern "C" void kernel_launch(void* const* d_in, const int* in_sizes, int n_in,
                              void* d_out, int out_size, void* d_ws, size_t ws_size,
                              hipStream_t stream) {
    const float* x        = (const float*)d_in[0];
    const float* ln1_w    = (const float*)d_in[1];
    const float* ln1_b    = (const float*)d_in[2];
    const float* in_proj  = (const float*)d_in[3];
    const float* conv_w   = (const float*)d_in[4];
    const float* conv_b   = (const float*)d_in[5];
    const float* x_proj   = (const float*)d_in[6];
    const float* dt_proj  = (const float*)d_in[7];
    const float* dt_b     = (const float*)d_in[8];
    const float* A_log    = (const float*)d_in[9];
    const float* Dp       = (const float*)d_in[10];
    const float* out_proj = (const float*)d_in[11];
    const float* ln2_w    = (const float*)d_in[12];
    const float* ln2_b    = (const float*)d_in[13];
    const float* mlp_w1   = (const float*)d_in[14];
    const float* mlp_b1   = (const float*)d_in[15];
    const float* mlp_w2   = (const float*)d_in[16];
    const float* mlp_b2   = (const float*)d_in[17];
    float* out = (float*)d_out;

    // ---- carve workspace ----
    char* base = (char*)d_ws;
    size_t off = 0;
    auto carve = [&](size_t bytes) -> char* {
        char* p = base + off;
        off = (off + bytes + 255) & ~(size_t)255;
        return p;
    };
    _Float16* w_in16  = (_Float16*)carve((size_t)2 * D_INNER * D_MODEL * 2);  // 4096x1024
    _Float16* w_xp16  = (_Float16*)carve((size_t)DBC_N * D_INNER * 2);        // 96x2048
    _Float16* w_dt16  = (_Float16*)carve((size_t)D_INNER * DT_RANK * 2);      // 2048x64
    _Float16* w_out16 = (_Float16*)carve((size_t)D_MODEL * D_INNER * 2);      // 1024x2048
    _Float16* w_m116  = (_Float16*)carve((size_t)D_INNER * D_MODEL * 2);      // 2048x1024
    _Float16* w_m216  = (_Float16*)carve((size_t)D_MODEL * D_INNER * 2);      // 1024x2048
    _Float16* xn16    = (_Float16*)carve((size_t)NTOK * D_MODEL * 2);
    float*    xz      = (float*)   carve((size_t)NTOK * 2 * D_INNER * 4);     // 32MB
    float*    u32     = (float*)   carve((size_t)NTOK * D_INNER * 4);
    _Float16* u16     = (_Float16*)carve((size_t)NTOK * D_INNER * 2);
    float*    dbc32   = (float*)   carve((size_t)NTOK * DBC_N * 4);
    _Float16* dbc16   = (_Float16*)carve((size_t)NTOK * DBC_N * 2);
    float*    delta   = (float*)   carve((size_t)NTOK * D_INNER * 4);
    float*    yscan   = (float*)   carve((size_t)NTOK * D_INNER * 4);
    _Float16* y16     = (_Float16*)carve((size_t)NTOK * D_INNER * 2);
    float*    mo      = (float*)   carve((size_t)NTOK * D_MODEL * 4);
    float*    hbuf    = (float*)   carve((size_t)NTOK * D_MODEL * 4);
    _Float16* hn16    = (_Float16*)carve((size_t)NTOK * D_MODEL * 2);
    // aliases for dead buffers:
    float*    mid     = xz;                 // xz dead after gate_silu
    _Float16* mid16   = u16;                // u16 dead after x_proj GEMM
    float*    y2      = u32;                // u32 dead after scan
    (void)ws_size; (void)n_in; (void)in_sizes; (void)out_size;

    const int T256 = 256;
    auto blocks = [](int n) { return (n + 255) / 256; };

    // 1) weights -> f16
    { int n = 2 * D_INNER * D_MODEL; cvt_f32_f16<<<blocks(n), T256, 0, stream>>>(in_proj,  w_in16,  n); }
    { int n = DBC_N * D_INNER;       cvt_f32_f16<<<blocks(n), T256, 0, stream>>>(x_proj,   w_xp16,  n); }
    { int n = D_INNER * DT_RANK;     cvt_f32_f16<<<blocks(n), T256, 0, stream>>>(dt_proj,  w_dt16,  n); }
    { int n = D_MODEL * D_INNER;     cvt_f32_f16<<<blocks(n), T256, 0, stream>>>(out_proj, w_out16, n); }
    { int n = D_INNER * D_MODEL;     cvt_f32_f16<<<blocks(n), T256, 0, stream>>>(mlp_w1,   w_m116,  n); }
    { int n = D_MODEL * D_INNER;     cvt_f32_f16<<<blocks(n), T256, 0, stream>>>(mlp_w2,   w_m216,  n); }

    // 2) LN1
    layernorm_f16<<<NTOK, T256, 0, stream>>>(x, ln1_w, ln1_b, xn16, D_MODEL);

    // 3) in_proj: xz = xn @ in_proj^T   (2048 x 4096, K=1024)
    gemm_wmma_f16<4><<<dim3((2 * D_INNER) / 64, NTOK / 16), 32, 0, stream>>>(
        xn16, D_MODEL, w_in16, D_MODEL, xz, 2 * D_INNER, D_MODEL);

    // 4) depthwise causal conv + SiLU -> u
    { int n = NTOK * D_INNER; conv_silu<<<blocks(n), T256, 0, stream>>>(xz, conv_w, conv_b, u32, u16, n); }

    // 5) x_proj: dbc = u @ x_proj^T    (2048 x 96, K=2048), 96 = 3 blocks of 2 tiles
    gemm_wmma_f16<2><<<dim3(DBC_N / 32, NTOK / 16), 32, 0, stream>>>(
        u16, D_INNER, w_xp16, D_INNER, dbc32, DBC_N, D_INNER);
    { int n = NTOK * DBC_N; cvt_f32_f16<<<blocks(n), T256, 0, stream>>>(dbc32, dbc16, n); }

    // 6) dt_proj: delta_raw = dt @ dt_proj^T  (2048 x 2048, K=64); dt = dbc[:, :64]
    gemm_wmma_f16<4><<<dim3(D_INNER / 64, NTOK / 16), 32, 0, stream>>>(
        dbc16, DBC_N, w_dt16, DT_RANK, delta, D_INNER, DT_RANK);

    // 7) softplus(+bias) in place
    { int n = NTOK * D_INNER; softplus_bias<<<blocks(n), T256, 0, stream>>>(delta, dt_b, n); }

    // 8) selective scan (includes + u*D)
    scan_kernel<<<(BATCH * D_INNER * D_STATE) / 256, T256, 0, stream>>>(
        delta, u32, dbc32, A_log, Dp, yscan);

    // 9) y = yscan * silu(z) -> f16
    { int n = NTOK * D_INNER; gate_silu<<<blocks(n), T256, 0, stream>>>(yscan, xz, y16, n); }

    // 10) out_proj: mo = y @ out_proj^T  (2048 x 1024, K=2048)
    gemm_wmma_f16<4><<<dim3(D_MODEL / 64, NTOK / 16), 32, 0, stream>>>(
        y16, D_INNER, w_out16, D_INNER, mo, D_MODEL, D_INNER);

    // 11) residual h = x + mamba_out
    { int n = NTOK * D_MODEL; add_kernel<<<blocks(n), T256, 0, stream>>>(x, mo, hbuf, n); }

    // 12) LN2
    layernorm_f16<<<NTOK, T256, 0, stream>>>(hbuf, ln2_w, ln2_b, hn16, D_MODEL);

    // 13) mlp1: mid = hn @ w1^T  (2048 x 2048, K=1024)
    gemm_wmma_f16<4><<<dim3(D_INNER / 64, NTOK / 16), 32, 0, stream>>>(
        hn16, D_MODEL, w_m116, D_MODEL, mid, D_INNER, D_MODEL);

    // 14) bias + exact GELU -> f16
    { int n = NTOK * D_INNER; bias_gelu<<<blocks(n), T256, 0, stream>>>(mid, mlp_b1, mid16, n); }

    // 15) mlp2: y2 = g @ w2^T  (2048 x 1024, K=2048)
    gemm_wmma_f16<4><<<dim3(D_MODEL / 64, NTOK / 16), 32, 0, stream>>>(
        mid16, D_INNER, w_m216, D_INNER, y2, D_MODEL, D_INNER);

    // 16) out = h + y2 + b2
    { int n = NTOK * D_MODEL; final_add<<<blocks(n), T256, 0, stream>>>(hbuf, y2, mlp_b2, out, n); }
}
